// AHT_22110491640672
// MI455X (gfx1250) — matrix operand order repaired and checked by reading, not compiled, not run on hardware
//
#include <hip/hip_runtime.h>
#include <hip/hip_bf16.h>
#include <math.h>

// Problem constants (from reference)
#define S_  512
#define K_  512
#define Q_  64
#define D_  256
#define H_  8
#define DH_ 32

typedef __attribute__((ext_vector_type(16))) __bf16 v16bf;
typedef __attribute__((ext_vector_type(8)))  float  v8f;
typedef __attribute__((ext_vector_type(2)))  __bf16 bf16x2;
typedef __attribute__((ext_vector_type(4)))  unsigned int u32x4;
typedef __attribute__((ext_vector_type(8)))  int i32x8;
typedef __attribute__((ext_vector_type(4)))  int i32x4;

// ---------------------------------------------------------------------------
// WMMA fragment helpers (wave32, V_WMMA_F32_16X16X32_BF16)
// A: 16x32 bf16. Lane l: row M = l&15. Lanes 0-15 hold K = {0..7,16..23},
// lanes 16-31 hold K = {8..15,24..31}; 2 bf16 per VGPR.
// B: 32x16 bf16. Lane l: col N = l&15. Lanes 0-15 hold K = {0..7,16..23}, lanes 16-31
// hold K = {8..15,24..31}; 2 bf16 per VGPR.
// C/D: 8 f32 VGPRs. Lane l: col N = l&15, VGPR r: row M = r + 8*(l>=16).
// ---------------------------------------------------------------------------
__device__ __forceinline__ v8f wmma_bf16(v16bf a, v16bf b, v8f c) {
  return __builtin_amdgcn_wmma_f32_16x16x32_bf16(
      /*neg_a=*/false, a, /*neg_b=*/false, b,
      /*c_mod=*/(short)0, c, /*reuse_a=*/false, /*reuse_b=*/false);
}

// A fragment from row-major bf16 tile (element (m,k) at src[m*ld + k])
__device__ __forceinline__ v16bf load_frag_a(const __bf16* src, int ld, int lane) {
  const int m  = lane & 15;
  const int kb = (lane >> 4) << 3;          // 0 or 8
  const __bf16* r = src + (size_t)m * ld;
  v16bf a;
#pragma unroll
  for (int v = 0; v < 8; ++v) {
    const int k = ((v < 4) ? 0 : 16) + kb + ((v & 3) << 1);
    const bf16x2 p = *(const bf16x2*)(r + k);
    a[2 * v] = p.x; a[2 * v + 1] = p.y;
  }
  return a;
}

// B fragment where B[k][n] = W[n][k] (K contiguous in memory)
__device__ __forceinline__ v16bf load_frag_b_trans(const __bf16* w, int ld, int lane) {
  const int n  = lane & 15;
  const int kb = (lane >> 4) << 3;          // 0 or 8
  const __bf16* r = w + (size_t)n * ld + kb;
  v16bf b;
#pragma unroll
  for (int v = 0; v < 8; ++v) {
    const bf16x2 p = *(const bf16x2*)(r + ((v < 4) ? 0 : 16) + ((v & 3) << 1));
    b[2 * v] = p.x; b[2 * v + 1] = p.y;
  }
  return b;
}

__device__ __forceinline__ float wave_sum(float v) {
#pragma unroll
  for (int m = 16; m > 0; m >>= 1) v += __shfl_xor(v, m, 32);
  return v;
}
__device__ __forceinline__ float wave_max(float v) {
#pragma unroll
  for (int m = 16; m > 0; m >>= 1) v = fmaxf(v, __shfl_xor(v, m, 32));
  return v;
}

#define V8F_ZERO {0.f,0.f,0.f,0.f,0.f,0.f,0.f,0.f}

// ---------------------------------------------------------------------------
// Tensor Data Mover: async 2D bf16 tile load Global -> LDS.
// D# per CDNA5 ISA:
//   group0: [1:0]=count=1, [63:32]=lds_addr, [120:64]=global_addr, [127:126]=2
//   group1: [17:16]=data_size(1 => 2B), [79:48]=tensor_dim0, [111:80]=tensor_dim1,
//           [127:112]=tile_dim0, [143:128]=tile_dim1, [207:160]=tensor_dim0_stride
//   groups 2/3: zero (2-D tensor)
// Issue from one wave only (TDM ignores EXEC; branch is wave-uniform).
// ---------------------------------------------------------------------------
__device__ __forceinline__ void tdm_load_tile_bf16(
    unsigned lds_off, const void* gaddr,
    unsigned tile_w, unsigned tile_h, unsigned row_stride) {
  const unsigned long long ga = (unsigned long long)(size_t)gaddr;
  u32x4 g0;
  g0[0] = 1u;                                        // count=1 (valid user D#)
  g0[1] = lds_off;                                   // lds_addr (bytes)
  g0[2] = (unsigned)(ga & 0xFFFFFFFFu);              // global_addr[31:0]
  g0[3] = (unsigned)((ga >> 32) & 0x1FFFFFFu) | (2u << 30);  // addr[56:32], type=2
  const unsigned td0 = 1u << 20, td1 = 1u << 20;     // generous tensor extents
  i32x8 g1;
  g1[0] = (int)(1u << 16);                           // data_size = 2 bytes
  g1[1] = (int)((td0 & 0xFFFFu) << 16);              // tensor_dim0[15:0] @ bits 63:48
  g1[2] = (int)((td0 >> 16) | ((td1 & 0xFFFFu) << 16));  // dim0 hi | dim1 lo
  g1[3] = (int)((td1 >> 16) | (tile_w << 16));       // dim1 hi | tile_dim0
  g1[4] = (int)(tile_h & 0xFFFFu);                   // tile_dim1 (tile_dim2=0)
  g1[5] = (int)row_stride;                           // tensor_dim0_stride[31:0]
  g1[6] = 0; g1[7] = 0;
  const i32x4 z4 = {0, 0, 0, 0};
#if __clang_major__ >= 23
  const i32x8 z8 = {0, 0, 0, 0, 0, 0, 0, 0};
  __builtin_amdgcn_tensor_load_to_lds(g0, g1, z4, z4, z8, 0);
#else
  __builtin_amdgcn_tensor_load_to_lds(g0, g1, z4, z4, 0);
#endif
}

// ---------------------------------------------------------------------------
// K0: f32 -> bf16 weight conversion (weights stay hot in L2: 4 x 128KB)
// ---------------------------------------------------------------------------
__global__ void aht_cvt_bf16(const float* __restrict__ src,
                             __bf16* __restrict__ dst, int n) {
  int i = blockIdx.x * blockDim.x + threadIdx.x;
  if (i < n) dst[i] = (__bf16)src[i];
}

// ---------------------------------------------------------------------------
// K_inv: inv[s][k] = q if add_ids[s][q]==k else -1
// ---------------------------------------------------------------------------
__global__ void aht_inv(const int* __restrict__ add_ids, int* __restrict__ inv) {
  const int s = blockIdx.x, t = threadIdx.x;
  inv[s * K_ + t] = -1;
  __syncthreads();
  if (t < Q_) inv[s * K_ + add_ids[s * Q_ + t]] = t;
}

// ---------------------------------------------------------------------------
// K1: Kp = key_t @ Wk^T + bk (row-major bf16, for QK^T trans-B loads)
//     Vt = (key_t @ Wv^T + bv)^T  ([s][d][k] bf16, so AV B-frags are
//     contiguous-in-k and vectorize to b128 loads)
// grid (K/64, S), 256 threads. 64x256 gathered bf16 tile in LDS.
// ---------------------------------------------------------------------------
__global__ __launch_bounds__(256) void aht_proj_kv(
    const float* __restrict__ x, const int* __restrict__ target_ids,
    const __bf16* __restrict__ wk, const float* __restrict__ bk,
    const __bf16* __restrict__ wv, const float* __restrict__ bv,
    __bf16* __restrict__ Kp, __bf16* __restrict__ Vt) {
  const int s = blockIdx.y, k0 = blockIdx.x * 64;
  const int t = threadIdx.x, lane = t & 31, w = t >> 5;

  __shared__ int    rows[64];
  __shared__ __bf16 Xs[64][264];            // padded row stride (bank spread)

  if (t < 64) rows[t] = target_ids[s * K_ + k0 + t];
  __syncthreads();
  for (int i = t; i < 4096; i += 256) {     // 64 rows * 64 float4
    const int r = i >> 6, c4 = (i & 63) << 2;
    const float4 v = *(const float4*)(x + (size_t)rows[r] * D_ + c4);
    Xs[r][c4 + 0] = (__bf16)v.x; Xs[r][c4 + 1] = (__bf16)v.y;
    Xs[r][c4 + 2] = (__bf16)v.z; Xs[r][c4 + 3] = (__bf16)v.w;
  }
  __syncthreads();

  for (int ww = 0; ww < 2; ++ww) {
    const __bf16* wmat = ww ? wv : wk;
    const float*  bias = ww ? bv : bk;
    for (int tile = w; tile < 64; tile += 8) {     // 4 M-tiles x 16 N-tiles
      const int mt = tile >> 4, nt = tile & 15;
      v8f c = V8F_ZERO;
#pragma unroll
      for (int kk = 0; kk < D_; kk += 32) {
        const v16bf a = load_frag_a(&Xs[mt * 16][kk], 264, lane);
        const v16bf b = load_frag_b_trans(wmat + (size_t)(nt * 16) * D_ + kk, D_, lane);
        c = wmma_bf16(a, b, c);
      }
      const int n = lane & 15, mo = (lane >> 4) << 3;
      const int col = nt * 16 + n;
      const float bb = bias[col];
      if (ww == 0) {
        __bf16* dst = Kp + ((size_t)s * K_ + k0) * D_;
#pragma unroll
        for (int r = 0; r < 8; ++r)
          dst[(size_t)(mt * 16 + mo + r) * D_ + col] = (__bf16)(c[r] + bb);
      } else {
        // transposed store: 8 consecutive k per lane -> single 16B store
        __bf16* dst = Vt + ((size_t)s * D_ + col) * K_ + k0 + mt * 16 + mo;
#pragma unroll
        for (int r = 0; r < 8; ++r)
          dst[r] = (__bf16)(c[r] + bb);
      }
    }
  }
}

// ---------------------------------------------------------------------------
// K2: Qp = query @ Wq^T + bq ; store f32 (residual) + bf16 (WMMA operand)
// grid S, 256 threads (M = Q = 64)
// ---------------------------------------------------------------------------
__global__ __launch_bounds__(256) void aht_proj_q(
    const float* __restrict__ x, const int* __restrict__ target_ids,
    const int* __restrict__ add_ids,
    const __bf16* __restrict__ wq, const float* __restrict__ bq,
    float* __restrict__ Qf, __bf16* __restrict__ Qb) {
  const int s = blockIdx.x;
  const int t = threadIdx.x, lane = t & 31, w = t >> 5;

  __shared__ int    rows[64];
  __shared__ __bf16 Xs[64][264];

  if (t < 64) rows[t] = target_ids[s * K_ + add_ids[s * Q_ + t]];
  __syncthreads();
  for (int i = t; i < 4096; i += 256) {
    const int r = i >> 6, c4 = (i & 63) << 2;
    const float4 v = *(const float4*)(x + (size_t)rows[r] * D_ + c4);
    Xs[r][c4 + 0] = (__bf16)v.x; Xs[r][c4 + 1] = (__bf16)v.y;
    Xs[r][c4 + 2] = (__bf16)v.z; Xs[r][c4 + 3] = (__bf16)v.w;
  }
  __syncthreads();

  for (int tile = w; tile < 64; tile += 8) {
    const int mt = tile >> 4, nt = tile & 15;
    v8f c = V8F_ZERO;
#pragma unroll
    for (int kk = 0; kk < D_; kk += 32) {
      const v16bf a = load_frag_a(&Xs[mt * 16][kk], 264, lane);
      const v16bf b = load_frag_b_trans(wq + (size_t)(nt * 16) * D_ + kk, D_, lane);
      c = wmma_bf16(a, b, c);
    }
    const int n = lane & 15, mo = (lane >> 4) << 3;
    const float bb = bq[nt * 16 + n];
#pragma unroll
    for (int r = 0; r < 8; ++r) {
      const float o = c[r] + bb;
      const size_t idx = ((size_t)s * Q_ + mt * 16 + mo + r) * D_ + nt * 16 + n;
      Qf[idx] = o; Qb[idx] = (__bf16)o;
    }
  }
}

// ---------------------------------------------------------------------------
// K3: attention per (h, s). Q-tile fetched by the Tensor Data Mover
// (64x32 bf16 tile, row stride D) into LDS; scores = Qp_h @ Kp_h^T / 16
// (one K=32 WMMA per tile since DH==32), softmax over K in 196KB dynamic
// LDS, then A @ V via the transposed Vt layout (contiguous-k B-frags).
// ---------------------------------------------------------------------------
__global__ __launch_bounds__(256) void aht_attn(
    const __bf16* __restrict__ Kp, const __bf16* __restrict__ Vt,
    const __bf16* __restrict__ Qb, const float* __restrict__ Qf,
    float* __restrict__ O1) {
  const int h = blockIdx.x, s = blockIdx.y;
  const int t = threadIdx.x, lane = t & 31, w = t >> 5;

  extern __shared__ char smem[];
  __bf16* Qs = (__bf16*)smem;                                // 64x32 bf16  (4KB)
  float*  Sc = (float*)(smem + 64 * 32 * 2);                 // 64x512 f32  (128KB)
  __bf16* P  = (__bf16*)(smem + 64 * 32 * 2 + 64 * 512 * 4); // 64x512 bf16 (64KB)

  // Async TDM tile load: wave 0 issues, TENSORcnt guards, barrier publishes.
  if (w == 0) {
    tdm_load_tile_bf16((unsigned)(size_t)Qs,
                       Qb + (size_t)s * Q_ * D_ + h * DH_,
                       /*tile_w=*/DH_, /*tile_h=*/Q_, /*row_stride=*/D_);
    __builtin_amdgcn_s_wait_tensorcnt(0);
  }
  __syncthreads();

  const __bf16* Kb = Kp + (size_t)s * K_ * D_ + h * DH_;
  for (int tile = w; tile < 128; tile += 8) {    // 4 M-tiles x 32 N-tiles
    const int mt = tile >> 5, nt = tile & 31;
    const v16bf a = load_frag_a(Qs + mt * 16 * 32, 32, lane);
    const v16bf b = load_frag_b_trans(Kb + (size_t)(nt * 16) * D_, D_, lane);
    v8f c = V8F_ZERO;
    c = wmma_bf16(a, b, c);
    const int n = lane & 15, mo = (lane >> 4) << 3;
#pragma unroll
    for (int r = 0; r < 8; ++r)
      Sc[(mt * 16 + mo + r) * 512 + nt * 16 + n] = c[r] * 0.0625f; // 1/sqrt(256)
  }
  __syncthreads();

  for (int row = w * 8; row < w * 8 + 8; ++row) {  // 8 rows per wave
    float* Sr = Sc + row * 512;
    float m = -3.4e38f;
    for (int j = lane; j < 512; j += 32) m = fmaxf(m, Sr[j]);
    m = wave_max(m);
    float sum = 0.f;
    for (int j = lane; j < 512; j += 32) { const float e = __expf(Sr[j] - m); Sr[j] = e; sum += e; }
    sum = wave_sum(sum);
    const float is = 1.f / sum;
    for (int j = lane; j < 512; j += 32) P[row * 512 + j] = (__bf16)(Sr[j] * is);
  }
  __syncthreads();

  {                                             // A@V: one 16x16 tile per wave
    const int mt = w >> 1, nt = w & 1;
    v8f c = V8F_ZERO;
    const __bf16* Vb = Vt + ((size_t)s * D_ + h * DH_ + nt * 16) * K_;
#pragma unroll
    for (int ks = 0; ks < 16; ++ks) {
      const v16bf a = load_frag_a(P + mt * 16 * 512 + ks * 32, 512, lane);
      const v16bf b = load_frag_b_trans(Vb + ks * 32, K_, lane);
      c = wmma_bf16(a, b, c);
    }
    const int n = lane & 15, mo = (lane >> 4) << 3;
#pragma unroll
    for (int r = 0; r < 8; ++r) {
      const size_t idx = ((size_t)s * Q_ + mt * 16 + mo + r) * D_ + h * DH_ + nt * 16 + n;
      O1[idx] = Qf[idx] + c[r];
    }
  }
}

// ---------------------------------------------------------------------------
// K4: LN1 -> z = y @ Wl^T + bl -> u = y + relu(z) -> LN2. grid S.
// ---------------------------------------------------------------------------
__global__ __launch_bounds__(256) void aht_ffn(
    const float* __restrict__ O1,
    const float* __restrict__ g1, const float* __restrict__ b1,
    const __bf16* __restrict__ wl, const float* __restrict__ bl,
    const float* __restrict__ g2, const float* __restrict__ b2,
    float* __restrict__ O2) {
  const int s = blockIdx.x;
  const int t = threadIdx.x, lane = t & 31, w = t >> 5;

  extern __shared__ char smem[];
  float*  Y  = (float*)smem;                       // 64x256 f32 (64KB)
  float*  U  = (float*)(smem + 64 * 256 * 4);      // 64x256 f32 (64KB)
  __bf16* Yb = (__bf16*)(smem + 2 * 64 * 256 * 4); // 64x256 bf16 (32KB)

  for (int row = w * 8; row < w * 8 + 8; ++row) {
    const float* src = O1 + ((size_t)s * Q_ + row) * D_;
    float sum = 0.f, sq = 0.f;
    for (int j = lane; j < D_; j += 32) { const float v = src[j]; sum += v; sq += v * v; }
    sum = wave_sum(sum); sq = wave_sum(sq);
    const float mu = sum * (1.f / D_);
    const float rs = rsqrtf(sq * (1.f / D_) - mu * mu + 1e-5f);
    for (int j = lane; j < D_; j += 32) {
      const float y = (src[j] - mu) * rs * g1[j] + b1[j];
      Y[row * D_ + j] = y; Yb[row * D_ + j] = (__bf16)y;
    }
  }
  __syncthreads();

  for (int tile = w; tile < 64; tile += 8) {
    const int mt = tile >> 4, nt = tile & 15;
    v8f c = V8F_ZERO;
#pragma unroll
    for (int kk = 0; kk < D_; kk += 32) {
      const v16bf a = load_frag_a(Yb + mt * 16 * D_ + kk, D_, lane);
      const v16bf b = load_frag_b_trans(wl + (size_t)(nt * 16) * D_ + kk, D_, lane);
      c = wmma_bf16(a, b, c);
    }
    const int n = lane & 15, mo = (lane >> 4) << 3;
    const float bb = bl[nt * 16 + n];
#pragma unroll
    for (int r = 0; r < 8; ++r) {
      const int m = mt * 16 + mo + r, col = nt * 16 + n;
      const float z = c[r] + bb;
      U[m * D_ + col] = Y[m * D_ + col] + fmaxf(z, 0.f);
    }
  }
  __syncthreads();

  for (int row = w * 8; row < w * 8 + 8; ++row) {
    const float* src = U + row * D_;
    float sum = 0.f, sq = 0.f;
    for (int j = lane; j < D_; j += 32) { const float v = src[j]; sum += v; sq += v * v; }
    sum = wave_sum(sum); sq = wave_sum(sq);
    const float mu = sum * (1.f / D_);
    const float rs = rsqrtf(sq * (1.f / D_) - mu * mu + 1e-5f);
    for (int j = lane; j < D_; j += 32)
      O2[((size_t)s * Q_ + row) * D_ + j] = (src[j] - mu) * rs * g2[j] + b2[j];
  }
}

// ---------------------------------------------------------------------------
// K5: out[s,k,:] = x[target_ids[s,k],:] + (inv[s,k]>=0 ? O2[s,inv[s,k],:] : 0)
// float4 streaming: 256MB write, 23.3 TB/s bound.
// ---------------------------------------------------------------------------
__global__ __launch_bounds__(256) void aht_fuse(
    const float* __restrict__ x, const int* __restrict__ target_ids,
    const int* __restrict__ inv, const float* __restrict__ O2,
    float* __restrict__ out) {
  const size_t i4 = (size_t)blockIdx.x * 256 + threadIdx.x;  // 16777216 total
  const int sk = (int)(i4 >> 6);
  const int c4 = (int)((i4 & 63) << 2);
  const int row = target_ids[sk];
  float4 v = *(const float4*)(x + (size_t)row * D_ + c4);
  const int q = inv[sk];
  if (q >= 0) {
    const int s = sk >> 9;
    const float4 o = *(const float4*)(O2 + ((size_t)s * Q_ + q) * D_ + c4);
    v.x += o.x; v.y += o.y; v.z += o.z; v.w += o.w;
  }
  *(float4*)(out + i4 * 4) = v;
}

// ---------------------------------------------------------------------------
extern "C" void kernel_launch(void* const* d_in, const int* in_sizes, int n_in,
                              void* d_out, int out_size, void* d_ws, size_t ws_size,
                              hipStream_t stream) {
  const float* x          = (const float*)d_in[0];
  const int*   target_ids = (const int*)d_in[1];
  const int*   add_ids    = (const int*)d_in[2];
  const float* Wq = (const float*)d_in[3];  const float* bq = (const float*)d_in[4];
  const float* Wk = (const float*)d_in[5];  const float* bk = (const float*)d_in[6];
  const float* Wv = (const float*)d_in[7];  const float* bv = (const float*)d_in[8];
  const float* g1 = (const float*)d_in[9];  const float* b1 = (const float*)d_in[10];
  const float* Wl = (const float*)d_in[11]; const float* bl = (const float*)d_in[12];
  const float* g2 = (const float*)d_in[13]; const float* b2 = (const float*)d_in[14];
  float* out = (float*)d_out;

  // Workspace carve-up
  char* p = (char*)d_ws;
  auto alloc = [&](size_t bytes) { char* r = p; p += (bytes + 255) & ~(size_t)255; return r; };
  __bf16* wqb = (__bf16*)alloc((size_t)D_ * D_ * 2);
  __bf16* wkb = (__bf16*)alloc((size_t)D_ * D_ * 2);
  __bf16* wvb = (__bf16*)alloc((size_t)D_ * D_ * 2);
  __bf16* wlb = (__bf16*)alloc((size_t)D_ * D_ * 2);
  __bf16* Kp  = (__bf16*)alloc((size_t)S_ * K_ * D_ * 2);   // 128 MB row-major
  __bf16* Vt  = (__bf16*)alloc((size_t)S_ * K_ * D_ * 2);   // 128 MB transposed
  __bf16* Qb  = (__bf16*)alloc((size_t)S_ * Q_ * D_ * 2);   // 16 MB
  float*  Qf  = (float*)alloc((size_t)S_ * Q_ * D_ * 4);    // 32 MB
  float*  O1  = (float*)alloc((size_t)S_ * Q_ * D_ * 4);    // 32 MB
  float*  O2  = (float*)alloc((size_t)S_ * Q_ * D_ * 4);    // 32 MB
  int*    inv = (int*)alloc((size_t)S_ * K_ * 4);           // 1 MB

  (void)in_sizes; (void)n_in; (void)out_size; (void)ws_size;

  // Large dynamic LDS (CDNA5 WGP has 320KB)
  hipFuncSetAttribute(reinterpret_cast<const void*>(aht_attn),
                      hipFuncAttributeMaxDynamicSharedMemorySize, 200704);
  hipFuncSetAttribute(reinterpret_cast<const void*>(aht_ffn),
                      hipFuncAttributeMaxDynamicSharedMemorySize, 163840);

  const int wN = D_ * D_;
  aht_cvt_bf16<<<wN / 256, 256, 0, stream>>>(Wq, wqb, wN);
  aht_cvt_bf16<<<wN / 256, 256, 0, stream>>>(Wk, wkb, wN);
  aht_cvt_bf16<<<wN / 256, 256, 0, stream>>>(Wv, wvb, wN);
  aht_cvt_bf16<<<wN / 256, 256, 0, stream>>>(Wl, wlb, wN);
  aht_inv<<<S_, 512, 0, stream>>>(add_ids, inv);

  aht_proj_kv<<<dim3(K_ / 64, S_), 256, 0, stream>>>(x, target_ids, wkb, bk, wvb, bv, Kp, Vt);
  aht_proj_q<<<S_, 256, 0, stream>>>(x, target_ids, add_ids, wqb, bq, Qf, Qb);
  aht_attn<<<dim3(H_, S_), 256, 200704, stream>>>(Kp, Vt, Qb, Qf, O1);
  aht_ffn<<<S_, 256, 163840, stream>>>(O1, g1, b1, wlb, bl, g2, b2, O2);
  aht_fuse<<<(S_ * K_ * D_ / 4) / 256, 256, 0, stream>>>(x, target_ids, inv, O2, out);
}
